// GRINGate_52544629899285
// MI455X (gfx1250) — matrix operand order repaired and checked by reading, not compiled, not run on hardware
//
#include <hip/hip_runtime.h>
#include <hip/hip_bf16.h>
#include <math.h>

// ---------- CDNA5 WMMA types ----------
typedef __bf16 v16bf __attribute__((ext_vector_type(16)));
typedef float  v8f   __attribute__((ext_vector_type(8)));

#define TOKENS      16384       // B*S
#define Dm          2048
#define Em          64
#define Hm          1024

// d_out layout (floats): idx[32768] scores[32768] probs[1048576] importance[64] load[64]
#define OUT_SCORES  32768
#define OUT_PROBS   65536
#define OUT_IMP     1114112
#define OUT_LOAD    1114176

// workspace layout: Wgb bf16 [64*2048], W1b bf16 [1024*2048], diff f32 [16384]
#define WGB_ELEMS   (Em * Dm)            // 131072
#define W1B_ELEMS   (Hm * Dm)            // 2097152
#define DIFF_BYTES_OFF  ((WGB_ELEMS + W1B_ELEMS) * 2)   // 4456448

// h1 kernel blocking
#define HM_TILE     64          // tokens per workgroup
#define NB_COLS     128         // N-block width
#define KC_LEN      128         // staged K-chunk length

// ---------------- fragment loaders (contiguous k = 16h + s, same in A and B) ----------------
// A: 16-bit 16x32 (MxK). Lanes 0-15 row m, K = 0..15 ; lanes 16-31: K = 16..31.
__device__ __forceinline__ v16bf load_a_frag(const __bf16* base, int rowStride,
                                             int k0, int lane) {
  v16bf a;
  const int m  = lane & 15;
  const int ko = (lane >> 4) ? 16 : 0;
  const __bf16* rowp = base + (size_t)m * rowStride + k0;
#pragma unroll
  for (int i = 0; i < 16; ++i) {
    const int v = i >> 1, h = i & 1;
    const int k = ko + 2 * v + h;
    a[i] = rowp[k];
  }
  return a;
}

// B: 16-bit 32x16 (KxN); weights stored row-major [n, k] with given stride.
// Lanes 0-15: col n = lane, K = 0..15 ; lanes 16-31: col n = lane-16, K = 16..31.
__device__ __forceinline__ v16bf load_b_frag(const __bf16* wrow0, int rowStride,
                                             int k0, int lane) {
  v16bf b;
  const int n  = lane & 15;
  const int ko = (lane >> 4) ? 16 : 0;
  const __bf16* p = wrow0 + (size_t)n * rowStride + k0 + ko;
#pragma unroll
  for (int i = 0; i < 16; ++i) b[i] = p[i];
  return b;
}

__device__ __forceinline__ v8f wmma_bf16(v16bf a, v16bf b, v8f c) {
  return __builtin_amdgcn_wmma_f32_16x16x32_bf16(false, a, false, b,
                                                 (short)0, c, false, false);
}

__device__ __forceinline__ float gelu_exact(float v) {
  return 0.5f * v * (1.0f + erff(v * 0.70710678118f));
}

// ---------------- kernel 0: zero importance/load accumulators ----------------
__global__ void grin_zero_kernel(float* out) {
  out[OUT_IMP + threadIdx.x] = 0.0f;   // 128 threads cover imp[64]+load[64]
}

// ---------------- kernel 1: f32 -> bf16 weight conversion ----------------
__global__ void grin_cvt_kernel(const float* Wg, const float* W1, __bf16* ws) {
  int i = blockIdx.x * blockDim.x + threadIdx.x;
  if (i < WGB_ELEMS) {
    ws[i] = (__bf16)Wg[i];
  } else if (i < WGB_ELEMS + W1B_ELEMS) {
    ws[i] = (__bf16)W1[i - WGB_ELEMS];
  }
}

// ---------------- kernel H: LN + h1 GEMM + GELU + W2 -> difficulty ----------------
// 64 tokens per WG; W1 staged through LDS so it is read exactly once per WG.
__global__ void __launch_bounds__(256)
grin_h1_kernel(const float* __restrict__ x,
               const __bf16* __restrict__ W1b,
               const float* __restrict__ gamma,
               const float* __restrict__ beta,
               const float* __restrict__ W2,
               float* __restrict__ diff_ws) {
  extern __shared__ char smem[];
  __bf16* hb  = (__bf16*)smem;                                  // 64 x 2048 (256 KB)
  __bf16* Bst = (__bf16*)(smem + HM_TILE * Dm * 2);             // 128 x 128 (32 KB)
  float*  fB  = (float*)(smem + HM_TILE * Dm * 2 + NB_COLS * KC_LEN * 2);
  float* red  = fB;          // 256
  float* red2 = fB + 256;    // 256
  float* muA  = fB + 512;    // 64
  float* rsA  = fB + 576;    // 64
  float* dacc = fB + 640;    // 64

  const int g    = blockIdx.x;
  const int tid  = threadIdx.x;
  const int lane = tid & 31;
  const int wave = tid >> 5;

  if (tid < 64) dacc[tid] = 0.0f;

  // ---- phase 1: load x tile (raw bf16 into hb), row stats ----
  const int r     = tid >> 2;        // 0..63 token row
  const int chunk = tid & 3;         // 4 chunks of 512 cols
  const float* xrow = x + (size_t)(g * HM_TILE + r) * Dm;
  const int c0 = chunk * 512;
  float s = 0.0f, s2 = 0.0f;
#pragma unroll 4
  for (int c = c0; c < c0 + 512; ++c) {
    float v = xrow[c];
    hb[r * Dm + c] = (__bf16)v;
    s += v; s2 += v * v;
  }
  red[tid] = s; red2[tid] = s2;
  __syncthreads();
  if (tid < 64) {
    float ss  = red[tid * 4] + red[tid * 4 + 1] + red[tid * 4 + 2] + red[tid * 4 + 3];
    float ss2 = red2[tid * 4] + red2[tid * 4 + 1] + red2[tid * 4 + 2] + red2[tid * 4 + 3];
    float mu  = ss * (1.0f / Dm);
    float var = ss2 * (1.0f / Dm) - mu * mu;
    muA[tid] = mu;
    rsA[tid] = rsqrtf(var + 1e-5f);
  }
  __syncthreads();

  // ---- phase 2: LayerNorm in place (bf16) ----
  {
    const float mu = muA[r], rs = rsA[r];
#pragma unroll 4
    for (int c = c0; c < c0 + 512; ++c) {
      float v = (float)hb[r * Dm + c];
      hb[r * Dm + c] = (__bf16)((v - mu) * rs * gamma[c] + beta[c]);
    }
  }
  __syncthreads();

  // ---- phase 3: blocked h1 = h @ W1^T with LDS-staged B ----
  const int mtile = wave >> 1;           // 0..3  (16-row tile)
  const int nhalf = wave & 1;            // 0..1  (64 of the 128 N-block cols)
  const __bf16* hbase = hb + (size_t)mtile * 16 * Dm;
  float part[8];
#pragma unroll
  for (int rr = 0; rr < 8; ++rr) part[rr] = 0.0f;

  for (int nb = 0; nb < Hm / NB_COLS; ++nb) {          // 8 N-blocks of 128
    v8f acc0 = {}, acc1 = {}, acc2 = {}, acc3 = {};
    for (int kc = 0; kc < Dm; kc += KC_LEN) {          // 16 staged K-chunks
      // cooperative stage: Bst[n][k] = W1b[nb*128+n][kc+k], 128x128 bf16
      {
        const int n  = tid >> 1;
        const int kh = tid & 1;                        // 64-elem half of the row
        const uint4* src = (const uint4*)(W1b + (size_t)(nb * NB_COLS + n) * Dm
                                          + kc + kh * 64);
        uint4* dst = (uint4*)(Bst + n * KC_LEN + kh * 64);
#pragma unroll
        for (int q = 0; q < 8; ++q) dst[q] = src[q];
      }
      __syncthreads();
#pragma unroll
      for (int ks = 0; ks < KC_LEN; ks += 32) {
        v16bf a = load_a_frag(hbase, Dm, kc + ks, lane);
        acc0 = wmma_bf16(a, load_b_frag(Bst + (nhalf * 64 +  0) * KC_LEN, KC_LEN, ks, lane), acc0);
        acc1 = wmma_bf16(a, load_b_frag(Bst + (nhalf * 64 + 16) * KC_LEN, KC_LEN, ks, lane), acc1);
        acc2 = wmma_bf16(a, load_b_frag(Bst + (nhalf * 64 + 32) * KC_LEN, KC_LEN, ks, lane), acc2);
        acc3 = wmma_bf16(a, load_b_frag(Bst + (nhalf * 64 + 48) * KC_LEN, KC_LEN, ks, lane), acc3);
      }
      __syncthreads();
    }
    // GELU + W2 dot for the 4 finished 16x16 tiles
    const int nlane = lane & 15;
    const int colbase = nb * NB_COLS + nhalf * 64 + nlane;
    const float w0 = W2[colbase +  0];
    const float w1 = W2[colbase + 16];
    const float w2c = W2[colbase + 32];
    const float w3 = W2[colbase + 48];
#pragma unroll
    for (int rr = 0; rr < 8; ++rr) {
      part[rr] += gelu_exact(acc0[rr]) * w0 + gelu_exact(acc1[rr]) * w1
                + gelu_exact(acc2[rr]) * w2c + gelu_exact(acc3[rr]) * w3;
    }
  }
  {
    const int m0 = mtile * 16 + ((lane < 16) ? 0 : 8);
#pragma unroll
    for (int rr = 0; rr < 8; ++rr) atomicAdd(&dacc[m0 + rr], part[rr]);
  }
  __syncthreads();
  if (tid < 64) diff_ws[(size_t)g * HM_TILE + tid] = dacc[tid];
}

// ---------------- kernel L: logits + softmax + top-2 + outputs ----------------
__global__ void __launch_bounds__(256)
grin_logits_kernel(const float* __restrict__ x,
                   const __bf16* __restrict__ Wgb,
                   const float* __restrict__ diff_ws,
                   float* __restrict__ out) {
  extern __shared__ char smem[];
  __bf16* xb   = (__bf16*)smem;                       // 16 x 2048 (64 KB)
  float*  logB = (float*)(smem + 16 * Dm * 2);        // 16 x 64
  float*  loadB = logB + 16 * Em;                     // 64

  const int g    = blockIdx.x;
  const int tid  = threadIdx.x;
  const int lane = tid & 31;
  const int wave = tid >> 5;

  for (int i = tid; i < 16 * Em; i += 256) logB[i] = 0.0f;
  if (tid < 64) loadB[tid] = 0.0f;

  // load x tile -> bf16 LDS
  {
    const int r = tid & 15, chunk = tid >> 4;
    const float* xrow = x + (size_t)(g * 16 + r) * Dm;
#pragma unroll 4
    for (int c = chunk * 128; c < chunk * 128 + 128; ++c)
      xb[r * Dm + c] = (__bf16)xrow[c];
  }
  __syncthreads();

  // logits: 8 waves = 4 E-tiles x 2 K-halves; combine via LDS atomics
  {
    const int e0    = (wave & 3) * 16;
    const int kbase = (wave >> 2) * (Dm / 2);
    v8f acc = {};
    for (int k0 = 0; k0 < Dm / 2; k0 += 32) {
      v16bf a = load_a_frag(xb, Dm, kbase + k0, lane);
      v16bf b = load_b_frag(Wgb + (size_t)e0 * Dm, Dm, kbase + k0, lane);
      acc = wmma_bf16(a, b, acc);
    }
    const int m0 = (lane < 16) ? 0 : 8;
    const int n  = lane & 15;
#pragma unroll
    for (int rr = 0; rr < 8; ++rr)
      atomicAdd(&logB[(m0 + rr) * Em + e0 + n], acc[rr]);
  }
  __syncthreads();

  // modulate, softmax, top-2, scores (one thread per token)
  if (tid < 16) {
    const int m = tid;
    const long t = (long)g * 16 + m;
    const float diff = 1.0f / (1.0f + __expf(-diff_ws[t]));  // sigmoid
    const float sc   = 1.0f / (1.0f + diff);                 // /(1+diff), TEMP=1
    float mx = -3.4e38f;
    for (int e = 0; e < Em; ++e) {
      float l = logB[m * Em + e] * sc;
      logB[m * Em + e] = l;
      mx = fmaxf(mx, l);
    }
    float sum = 0.0f;
    for (int e = 0; e < Em; ++e) {
      float p = __expf(logB[m * Em + e] - mx);
      logB[m * Em + e] = p;
      sum += p;
    }
    const float inv = 1.0f / sum;
    int i1 = 0, i2 = 1;
    float p1 = -1.0f, p2 = -1.0f;
    for (int e = 0; e < Em; ++e) {
      float p = logB[m * Em + e] * inv;
      logB[m * Em + e] = p;                      // keep probs for importance
      out[OUT_PROBS + t * Em + e] = p;
      if (p > p1)      { i2 = i1; p2 = p1; i1 = e; p1 = p; }
      else if (p > p2) { i2 = e;  p2 = p; }
    }
    float s1  = (1.0f - p1) + p1;                // straight-through, faithful fp
    float s2v = (1.0f - p2) + p2;
    float den = fmaxf(s1 + s2v, 1e-9f);
    out[t * 2 + 0] = (float)i1;
    out[t * 2 + 1] = (float)i2;
    out[OUT_SCORES + t * 2 + 0] = s1 / den;
    out[OUT_SCORES + t * 2 + 1] = s2v / den;
    atomicAdd(&loadB[i1], 1.0f);
    atomicAdd(&loadB[i2], 1.0f);
  }
  __syncthreads();

  if (tid < Em) {
    float se = 0.0f;
    for (int m = 0; m < 16; ++m) se += logB[m * Em + tid];
    atomicAdd(&out[OUT_IMP + tid],  se * (1.0f / TOKENS));
    atomicAdd(&out[OUT_LOAD + tid], loadB[tid] * (1.0f / TOKENS));
  }
}

extern "C" void kernel_launch(void* const* d_in, const int* in_sizes, int n_in,
                              void* d_out, int out_size, void* d_ws, size_t ws_size,
                              hipStream_t stream) {
  const float* x     = (const float*)d_in[0];
  const float* Wg    = (const float*)d_in[1];
  const float* gamma = (const float*)d_in[2];
  const float* beta  = (const float*)d_in[3];
  const float* W1    = (const float*)d_in[4];
  const float* W2    = (const float*)d_in[5];
  float* out = (float*)d_out;

  __bf16* Wgb = (__bf16*)d_ws;
  __bf16* W1b = Wgb + WGB_ELEMS;
  float*  diff_ws = (float*)((char*)d_ws + DIFF_BYTES_OFF);

  grin_zero_kernel<<<1, 128, 0, stream>>>(out);

  {
    int total = WGB_ELEMS + W1B_ELEMS;
    grin_cvt_kernel<<<(total + 255) / 256, 256, 0, stream>>>(Wg, W1, (__bf16*)d_ws);
  }

  // h1 pipeline: 64 tokens/WG, W1 read once per WG through LDS staging
  {
    size_t smem = (size_t)HM_TILE * Dm * 2 + (size_t)NB_COLS * KC_LEN * 2
                + (size_t)(256 + 256 + 64 + 64 + 64) * 4;
    hipFuncSetAttribute((const void*)grin_h1_kernel,
                        hipFuncAttributeMaxDynamicSharedMemorySize, (int)smem);
    grin_h1_kernel<<<TOKENS / HM_TILE, 256, smem, stream>>>(
        x, W1b, gamma, beta, W2, diff_ws);
  }

  // logits + softmax + outputs: 16 tokens/WG, all 8 waves on the E x K split
  {
    size_t smem = (size_t)16 * Dm * 2 + (size_t)(16 * Em + 64) * 4;
    hipFuncSetAttribute((const void*)grin_logits_kernel,
                        hipFuncAttributeMaxDynamicSharedMemorySize, (int)smem);
    grin_logits_kernel<<<TOKENS / 16, 256, smem, stream>>>(x, Wgb, diff_ws, out);
  }
}
